// Model_21363167330511
// MI455X (gfx1250) — matrix operand order repaired and checked by reading, not compiled, not run on hardware
//
#include <hip/hip_runtime.h>

typedef __attribute__((ext_vector_type(16))) _Float16 v16h;
typedef __attribute__((ext_vector_type(8)))  float    v8f;

static constexpr int NCN = 100000;   // customers
static constexpr int NPN = 50000;    // products
static constexpr int EE  = 500000;   // edges
static constexpr float SLOPE = 0.2f;

// ---------------------------------------------------------------------------
// Weight prep: B[K,N] f32 row-major  ->  Bt[N,K] f16 (transposed, contiguous-K)
// ---------------------------------------------------------------------------
__global__ void wprep(const float* __restrict__ B, _Float16* __restrict__ Bt,
                      int K, int N) {
  int idx = blockIdx.x * blockDim.x + threadIdx.x;
  if (idx >= K * N) return;
  int n = idx / K, k = idx % K;
  Bt[(size_t)n * K + k] = (_Float16)B[(size_t)k * N + n];
}

// ---------------------------------------------------------------------------
// WMMA GEMM: C[M, NT*16] = A[M,K] (f32 row-major) * W (= Bt[N,K] f16)
// Weights staged once per block into LDS (padded rows -> conflict-free
// ds_load_b128); A converted f32->f16 in-register; WMMA f32 accumulate.
// One wave per 16-row strip; A fragment reused across all NT N-tiles.
// ---------------------------------------------------------------------------
template <int NT, int K>
__global__ __launch_bounds__(128)
void gemm_wmma(const float* __restrict__ A, const _Float16* __restrict__ Bt,
               float* __restrict__ C, int M) {
  constexpr int N  = NT * 16;
  constexpr int KP = K + 8;                 // +16B row pad: banks shift 4/row
  __shared__ _Float16 Bs[N * KP];

  // ---- cooperative stage of Bt into LDS (16B chunks) ----
  {
    constexpr int CPR = K / 8;              // 16B chunks per row
    for (int i = threadIdx.x; i < N * CPR; i += 128) {
      int n = i / CPR, c = i % CPR;
      *reinterpret_cast<float4*>(&Bs[n * KP + c * 8]) =
          *reinterpret_cast<const float4*>(Bt + (size_t)n * K + c * 8);
    }
  }
  __syncthreads();                          // all waves participate, then exit

  const int wave  = threadIdx.x >> 5;
  const int lane  = threadIdx.x & 31;
  const int strip = blockIdx.x * 4 + wave;
  if (strip * 16 >= M) return;              // wave-uniform: EXEC stays all-ones
  const int hl  = lane >> 4;                // half-wave id (0/1)
  const int l16 = lane & 15;

  const float* Ar = A + (size_t)(strip * 16 + l16) * K;

  v8f zero = {};
  v8f acc[NT];
#pragma unroll
  for (int t = 0; t < NT; ++t) acc[t] = zero;

  for (int k0 = 0; k0 < K; k0 += 32) {
    if (k0 + 32 < K)  // prefetch next K-step of this A row (global_prefetch_b8)
      __builtin_prefetch(Ar + k0 + 32, 0, 1);
    // ---- A fragment (ISA 7.12.2 16-bit A 16x32 layout) ----
    const float4* pa0 = reinterpret_cast<const float4*>(Ar + k0 + hl * 8);
    const float4* pa1 = reinterpret_cast<const float4*>(Ar + k0 + 16 + hl * 8);
    float4 a0 = pa0[0], a1 = pa0[1];
    float4 a2 = pa1[0], a3 = pa1[1];
    v16h af;
    af[0]  = (_Float16)a0.x; af[1]  = (_Float16)a0.y;
    af[2]  = (_Float16)a0.z; af[3]  = (_Float16)a0.w;
    af[4]  = (_Float16)a1.x; af[5]  = (_Float16)a1.y;
    af[6]  = (_Float16)a1.z; af[7]  = (_Float16)a1.w;
    af[8]  = (_Float16)a2.x; af[9]  = (_Float16)a2.y;
    af[10] = (_Float16)a2.z; af[11] = (_Float16)a2.w;
    af[12] = (_Float16)a3.x; af[13] = (_Float16)a3.y;
    af[14] = (_Float16)a3.z; af[15] = (_Float16)a3.w;

#pragma unroll
    for (int t = 0; t < NT; ++t) {
      const int col = t * 16 + l16;
      // bf[j] = Bs[col, k0+hl*8+j] (j<8), Bs[col, k0+16+hl*8+(j-8)] (j>=8)
      // from LDS (2x ds_load_b128). Matches the A fragment k order.
      typedef __attribute__((ext_vector_type(8))) _Float16 v8h_;
      const v8h_ b0 = *reinterpret_cast<const v8h_*>(&Bs[col * KP + k0 + hl * 8]);
      const v8h_ b1 = *reinterpret_cast<const v8h_*>(&Bs[col * KP + k0 + 16 + hl * 8]);
      const v16h bf = __builtin_shufflevector(b0, b1, 0,1,2,3,4,5,6,7,8,9,10,11,12,13,14,15);
      acc[t] = __builtin_amdgcn_wmma_f32_16x16x32_f16(
          /*neg_a=*/false, af, /*neg_b=*/false, bf,
          /*c_mod=*/(short)0, acc[t], /*reuse_a=*/false, /*reuse_b=*/false);
    }
  }

#pragma unroll
  for (int t = 0; t < NT; ++t) {
#pragma unroll
    for (int i = 0; i < 8; ++i) {
      const int r = strip * 16 + hl * 8 + i;
      C[(size_t)r * N + t * 16 + l16] = acc[t][i];
    }
  }
}

// ---------------------------------------------------------------------------
// Attention helpers
// ---------------------------------------------------------------------------
__device__ __forceinline__ unsigned int encf(float x) {
  unsigned int u = __float_as_uint(x);
  return (u >> 31) ? ~u : (u | 0x80000000u);   // monotone float->uint map
}
__device__ __forceinline__ float decf(unsigned int u) {
  return (u >> 31) ? __uint_as_float(u ^ 0x80000000u) : __uint_as_float(~u);
}

// s[n,h] = sum_c h[n,h,c] * att[h,c]
__global__ void node_scores(const float* __restrict__ h, const float* __restrict__ att,
                            float* __restrict__ s, int N, int H, int C) {
  int idx = blockIdx.x * blockDim.x + threadIdx.x;
  if (idx >= N * H) return;
  int n = idx / H, hh = idx % H;
  const float4* hp = reinterpret_cast<const float4*>(h + (size_t)n * H * C + (size_t)hh * C);
  const float4* ap = reinterpret_cast<const float4*>(att + (size_t)hh * C);
  float acc = 0.f;
  for (int c = 0; c < C / 4; ++c) {
    float4 hv = hp[c], av = ap[c];
    acc += hv.x * av.x + hv.y * av.y + hv.z * av.z + hv.w * av.w;
  }
  s[idx] = acc;
}

// alpha = leaky_relu(sS[src] + sD[dst]); segment-max into menc (encoded uint)
__global__ void edge_alpha_max(const float* __restrict__ sS, const float* __restrict__ sD,
                               const int* __restrict__ esrc, const int* __restrict__ edst,
                               float* __restrict__ alpha, unsigned int* __restrict__ menc,
                               int E, int H) {
  int idx = blockIdx.x * blockDim.x + threadIdx.x;
  if (idx >= E * H) return;
  int e = idx / H, hh = idx % H;
  float a = sS[esrc[e] * H + hh] + sD[edst[e] * H + hh];
  a = (a > 0.f) ? a : SLOPE * a;
  alpha[idx] = a;
  atomicMax(&menc[edst[e] * H + hh], encf(a));
}

// ex = exp(alpha - m[dst]); segment-sum into den
__global__ void edge_exp_sum(const float* __restrict__ alpha, const int* __restrict__ edst,
                             const unsigned int* __restrict__ menc,
                             float* __restrict__ exb, float* __restrict__ den,
                             int E, int H) {
  int idx = blockIdx.x * blockDim.x + threadIdx.x;
  if (idx >= E * H) return;
  int e = idx / H, hh = idx % H;
  int d = edst[e];
  float ex = __expf(alpha[idx] - decf(menc[d * H + hh]));
  exb[idx] = ex;
  atomicAdd(&den[d * H + hh], ex);
}

// out[dst, f] += h_src[src, f] * ex/(den[dst]+eps)   (block = edge, thread = feature)
__global__ void edge_scatter(const float* __restrict__ hsrc,
                             const int* __restrict__ esrc, const int* __restrict__ edst,
                             const float* __restrict__ exb, const float* __restrict__ den,
                             float* __restrict__ out, int H, int C) {
  int e = blockIdx.x;
  int f = threadIdx.x;                 // f in [0, H*C)
  int hh = f / C;
  int s = esrc[e], d = edst[e];
  float w = exb[e * H + hh] / (den[d * H + hh] + 1e-16f);
  atomicAdd(&out[(size_t)d * H * C + f], hsrc[(size_t)s * H * C + f] * w);
}

__global__ void bias_act(float* __restrict__ out, const float* __restrict__ bias,
                         long long total, int F, int relu) {
  long long idx = (long long)blockIdx.x * blockDim.x + threadIdx.x;
  if (idx >= total) return;
  float v = out[idx] + bias[idx % F];
  if (relu) v = fmaxf(v, 0.f);
  out[idx] = v;
}

// ---------------------------------------------------------------------------
// Host-side conv driver (all launches on `stream`; graph-capture safe)
// ---------------------------------------------------------------------------
static void run_conv(const float* hS, const float* hD, int Ns, int Nd,
                     const float* attS, const float* attD,
                     const int* es, const int* ed, int H, int C,
                     float* out, const float* bias, int relu,
                     float* sS, float* sD, unsigned int* menc, float* den,
                     float* alpha, float* exb, hipStream_t stream) {
  hipMemsetAsync(menc, 0, (size_t)Nd * H * 4, stream);
  hipMemsetAsync(den,  0, (size_t)Nd * H * 4, stream);
  hipMemsetAsync(out,  0, (size_t)Nd * H * C * 4, stream);
  const int T = 256;
  node_scores<<<(Ns * H + T - 1) / T, T, 0, stream>>>(hS, attS, sS, Ns, H, C);
  node_scores<<<(Nd * H + T - 1) / T, T, 0, stream>>>(hD, attD, sD, Nd, H, C);
  edge_alpha_max<<<(EE * H + T - 1) / T, T, 0, stream>>>(sS, sD, es, ed, alpha, menc, EE, H);
  edge_exp_sum<<<(EE * H + T - 1) / T, T, 0, stream>>>(alpha, ed, menc, exb, den, EE, H);
  edge_scatter<<<EE, H * C, 0, stream>>>(hS, es, ed, exb, den, out, H, C);
  long long tot = (long long)Nd * H * C;
  bias_act<<<(unsigned)((tot + T - 1) / T), T, 0, stream>>>(out, bias, tot, H * C, relu);
}

extern "C" void kernel_launch(void* const* d_in, const int* in_sizes, int n_in,
                              void* d_out, int out_size, void* d_ws, size_t ws_size,
                              hipStream_t stream) {
  (void)in_sizes; (void)n_in; (void)out_size; (void)ws_size;
  const float* xc      = (const float*)d_in[0];
  const float* xp      = (const float*)d_in[1];
  const int*   esrc    = (const int*)d_in[2];
  const int*   edst    = (const int*)d_in[3];
  const float* w1b_src = (const float*)d_in[4];
  const float* w1b_dst = (const float*)d_in[5];
  const float* a1b_src = (const float*)d_in[6];
  const float* a1b_dst = (const float*)d_in[7];
  const float* b1b     = (const float*)d_in[8];
  const float* w1r_src = (const float*)d_in[9];
  const float* w1r_dst = (const float*)d_in[10];
  const float* a1r_src = (const float*)d_in[11];
  const float* a1r_dst = (const float*)d_in[12];
  const float* b1r     = (const float*)d_in[13];
  const float* w2b_src = (const float*)d_in[14];
  const float* w2b_dst = (const float*)d_in[15];
  const float* a2b_src = (const float*)d_in[16];
  const float* a2b_dst = (const float*)d_in[17];
  const float* b2b     = (const float*)d_in[18];
  const float* w2r_src = (const float*)d_in[19];
  const float* w2r_dst = (const float*)d_in[20];
  const float* a2r_src = (const float*)d_in[21];
  const float* a2r_dst = (const float*)d_in[22];
  const float* b2r     = (const float*)d_in[23];

  // ---- workspace bump allocator (256B aligned) ----
  char*  ws  = (char*)d_ws;
  size_t off = 0;
  auto alloc = [&](size_t bytes) -> void* {
    void* p = ws + off;
    off += (bytes + 255) & ~(size_t)255;
    return p;
  };
  float* h1b_src = (float*)alloc((size_t)NCN * 128 * 4);  // layer1 projections
  float* h1b_dst = (float*)alloc((size_t)NPN * 128 * 4);
  float* h1r_src = (float*)alloc((size_t)NPN * 128 * 4);
  float* h1r_dst = (float*)alloc((size_t)NCN * 128 * 4);
  float* c1      = (float*)alloc((size_t)NCN * 128 * 4);
  float* p1      = (float*)alloc((size_t)NPN * 128 * 4);
  float* sS      = (float*)alloc((size_t)NCN * 2 * 4);    // per-conv scratch
  float* sD      = (float*)alloc((size_t)NCN * 2 * 4);
  unsigned int* menc = (unsigned int*)alloc((size_t)NCN * 2 * 4);
  float* den     = (float*)alloc((size_t)NCN * 2 * 4);
  float* alpha   = (float*)alloc((size_t)EE * 2 * 4);
  float* exb     = (float*)alloc((size_t)EE * 2 * 4);
  // f16 transposed weights
  _Float16* wt1b_src = (_Float16*)alloc((size_t)256 * 128 * 2);
  _Float16* wt1b_dst = (_Float16*)alloc((size_t)128 * 128 * 2);
  _Float16* wt1r_src = (_Float16*)alloc((size_t)128 * 128 * 2);
  _Float16* wt1r_dst = (_Float16*)alloc((size_t)256 * 128 * 2);
  _Float16* wt2b_src = (_Float16*)alloc((size_t)128 * 64 * 2);
  _Float16* wt2b_dst = (_Float16*)alloc((size_t)128 * 64 * 2);
  _Float16* wt2r_src = (_Float16*)alloc((size_t)128 * 64 * 2);
  _Float16* wt2r_dst = (_Float16*)alloc((size_t)128 * 64 * 2);
  // layer-2 projections alias dead layer-1 buffers
  float* h2b_src = h1b_src;   // [NCN,64]
  float* h2b_dst = h1b_dst;   // [NPN,64]
  float* h2r_src = h1r_src;   // [NPN,64]
  float* h2r_dst = h1r_dst;   // [NCN,64]

  float* c2 = (float*)d_out;                       // [NCN,64]
  float* p2 = (float*)d_out + (size_t)NCN * 64;    // [NPN,64]

  // ---- weight transpose+convert (tiny) ----
  const int T = 256;
  wprep<<<(256 * 128 + T - 1) / T, T, 0, stream>>>(w1b_src, wt1b_src, 256, 128);
  wprep<<<(128 * 128 + T - 1) / T, T, 0, stream>>>(w1b_dst, wt1b_dst, 128, 128);
  wprep<<<(128 * 128 + T - 1) / T, T, 0, stream>>>(w1r_src, wt1r_src, 128, 128);
  wprep<<<(256 * 128 + T - 1) / T, T, 0, stream>>>(w1r_dst, wt1r_dst, 256, 128);
  wprep<<<(128 * 64 + T - 1) / T, T, 0, stream>>>(w2b_src, wt2b_src, 128, 64);
  wprep<<<(128 * 64 + T - 1) / T, T, 0, stream>>>(w2b_dst, wt2b_dst, 128, 64);
  wprep<<<(128 * 64 + T - 1) / T, T, 0, stream>>>(w2r_src, wt2r_src, 128, 64);
  wprep<<<(128 * 64 + T - 1) / T, T, 0, stream>>>(w2r_dst, wt2r_dst, 128, 64);

  const int gc = (NCN / 16 + 3) / 4;   // blocks for M=NCN strips (4 waves/block)
  const int gp = (NPN / 16 + 3) / 4;

  // ---- layer 1 projections (N=128 -> 8 tiles) ----
  gemm_wmma<8, 256><<<gc, 128, 0, stream>>>(xc, wt1b_src, h1b_src, NCN);
  gemm_wmma<8, 128><<<gp, 128, 0, stream>>>(xp, wt1b_dst, h1b_dst, NPN);
  gemm_wmma<8, 128><<<gp, 128, 0, stream>>>(xp, wt1r_src, h1r_src, NPN);
  gemm_wmma<8, 256><<<gc, 128, 0, stream>>>(xc, wt1r_dst, h1r_dst, NCN);

  // ---- layer 1 convs (H=2, C=64); bias+relu fused ----
  run_conv(h1b_src, h1b_dst, NCN, NPN, a1b_src, a1b_dst, esrc, edst, 2, 64,
           p1, b1b, /*relu=*/1, sS, sD, menc, den, alpha, exb, stream);
  run_conv(h1r_src, h1r_dst, NPN, NCN, a1r_src, a1r_dst, edst, esrc, 2, 64,
           c1, b1r, /*relu=*/1, sS, sD, menc, den, alpha, exb, stream);

  // ---- layer 2 projections (K=128, N=64 -> 4 tiles) ----
  gemm_wmma<4, 128><<<gc, 128, 0, stream>>>(c1, wt2b_src, h2b_src, NCN);
  gemm_wmma<4, 128><<<gp, 128, 0, stream>>>(p1, wt2b_dst, h2b_dst, NPN);
  gemm_wmma<4, 128><<<gp, 128, 0, stream>>>(p1, wt2r_src, h2r_src, NPN);
  gemm_wmma<4, 128><<<gc, 128, 0, stream>>>(c1, wt2r_dst, h2r_dst, NCN);

  // ---- layer 2 convs (H=1, C=64); bias only, outputs straight into d_out ----
  run_conv(h2b_src, h2b_dst, NCN, NPN, a2b_src, a2b_dst, esrc, edst, 1, 64,
           p2, b2b, /*relu=*/0, sS, sD, menc, den, alpha, exb, stream);
  run_conv(h2r_src, h2r_dst, NPN, NCN, a2r_src, a2r_dst, edst, esrc, 1, 64,
           c2, b2r, /*relu=*/0, sS, sD, menc, den, alpha, exb, stream);
}
